// AttentionNet_44839458570402
// MI455X (gfx1250) — matrix operand order repaired and checked by reading, not compiled, not run on hardware
//
#include <hip/hip_runtime.h>
#include <hip/hip_bf16.h>
#include <math.h>

// ---------------------------------------------------------------------------
// AttentionNet for MI455X (gfx1250, wave32, WMMA 16x16x32 f16 -> f32 acc)
//
//   pack_w      : f32 weights -> f16, pre-swizzled into WMMA B-fragment order
//   build_x_*   : gather f32 inputs -> padded f16 activation rows
//   mlp3        : fused 3-layer MLP; 128-row tiles, activations in LDS
//                 (~284KB/WGP — uses CDNA5's 320KB LDS), X tile staged by the
//                 Tensor Data Mover (tensor_load_to_lds + s_wait_tensorcnt)
//   attn        : per-batch-item attention/layernorm block, 1 wave / item
//   gemm64      : generic 64x64 WMMA GEMM tile for the value head
//   head_final  : 16x16 + 16x1 tail, per-row VALU
// ---------------------------------------------------------------------------

typedef __attribute__((ext_vector_type(16))) _Float16 v16h;
typedef __attribute__((ext_vector_type(8)))  _Float16 v8h;
typedef __attribute__((ext_vector_type(8)))  float    v8f;
typedef __attribute__((ext_vector_type(4)))  unsigned int u32x4;
typedef __attribute__((ext_vector_type(4)))  int i32x4;
typedef __attribute__((ext_vector_type(8)))  int i32x8;

#define BATCH 8192
#define NF 20
#define NA 4
#define ND 64
#define NH 512
#define HSTR 520   // LDS row stride (halves) for 512-wide planes: 16B skew -> no bank conflicts

#if defined(__HIP_DEVICE_COMPILE__) && __has_builtin(__builtin_amdgcn_tensor_load_to_lds)
#define HAVE_TDM 1
#else
#define HAVE_TDM 0
#endif

static __device__ __forceinline__ v8f wmma_f16(v16h a, v16h b, v8f c) {
  // (neg_a, A, neg_b, B, c_mod, C, reuse_a, reuse_b)
  return __builtin_amdgcn_wmma_f32_16x16x32_f16(false, a, false, b, (short)0, c,
                                                false, false);
}

// A fragment (16x32 f16, M = lane%16): 16 contiguous halves per lane at
// K offset lh*16 (matches the packed B order k = 16*g + kk).
static __device__ __forceinline__ v16h load_a(const _Float16* rowbase, int lh) {
  union { v16h v; v8h h[2]; } u;
  u.h[0] = *(const v8h*)(rowbase + lh * 16);
  u.h[1] = *(const v8h*)(rowbase + lh * 16 + 8);
  return u.v;
}

// B fragment from pre-swizzled weights: flat layout ((kt*N + n)*2 + g)*16 + kk,
// so a lane (n = n0+lm, g = lh) reads 16 contiguous halves (32 bytes).
static __device__ __forceinline__ v16h load_b(const _Float16* Wp, int N, int kt,
                                              int col, int lh) {
  return *(const v16h*)(Wp + ((size_t)(kt * N + col) * 2 + lh) * 16);
}
static __device__ __forceinline__ const _Float16* b_ptr(const _Float16* Wp, int N,
                                                        int kt, int col, int lh) {
  return Wp + ((size_t)(kt * N + col) * 2 + lh) * 16;
}

#if HAVE_TDM
// Issue one TDM 2D tile load (global -> LDS). Descriptor per CDNA5 ISA §8:
// group0 = {count=1 | lds_addr | global_addr | type=2}, group1 carries
// data_size=2B, tensor_dim0 = row width, tile = rows x width, dim0 stride.
// clang-23 toolchain form: 6 args (g0, g1, g2, g3, g4, cpol).
static __device__ __forceinline__ void tdm_load_2d(const void* gaddr,
                                                   unsigned lds_byte_addr,
                                                   unsigned width_elems,
                                                   unsigned rows,
                                                   unsigned row_stride_elems) {
  unsigned long long ga = (unsigned long long)gaddr;
  u32x4 g0;
  g0[0] = 1u;                                       // count=1 (valid user D#)
  g0[1] = lds_byte_addr;                            // lds_addr [63:32]
  g0[2] = (unsigned)(ga & 0xffffffffu);             // global_addr lo
  g0[3] = (unsigned)((ga >> 32) & 0x01ffffffu) | (2u << 30);  // addr hi | type=2
  i32x8 g1;
  g1[0] = (int)(1u << 16);                          // wg_mask=0, data_size=2B
  g1[1] = (int)((width_elems & 0xffffu) << 16);     // tensor_dim0 [15:0]
  g1[2] = (int)(((width_elems >> 16) & 0xffffu) |   // tensor_dim0 [31:16]
                ((rows & 0xffffu) << 16));          // tensor_dim1 [15:0]
  g1[3] = (int)(((rows >> 16) & 0xffffu) |          // tensor_dim1 [31:16]
                ((width_elems & 0xffffu) << 16));   // tile_dim0
  g1[4] = (int)(rows & 0xffffu);                    // tile_dim1 (tile_dim2=0)
  g1[5] = (int)row_stride_elems;                    // tensor_dim0_stride lo
  g1[6] = 0;                                        // stride hi / dim1_stride lo
  g1[7] = 0;
  i32x4 z4 = {0, 0, 0, 0};
  i32x8 z8 = {0, 0, 0, 0, 0, 0, 0, 0};
  __builtin_amdgcn_tensor_load_to_lds(g0, g1, z4, z4, z8, 0);
}
#endif

// ---------------------------------------------------------------------------
// Weight pre-swizzle: W (K x N row-major f32) -> f16 fragments, K padded to Kp
// (multiple of 32), N padded to Np (multiple of 16), zeros in the padding.
// ---------------------------------------------------------------------------
__global__ void pack_w_kernel(const float* __restrict__ W, int K, int N,
                              int Kp, int Np, _Float16* __restrict__ out) {
  size_t idx = (size_t)blockIdx.x * 256 + threadIdx.x;
  size_t total = (size_t)(Kp / 32) * Np * 32;
  if (idx >= total) return;
  int kk = (int)(idx & 15);
  int g  = (int)((idx >> 4) & 1);
  size_t cn = idx >> 5;          // kt*Np + n
  int n  = (int)(cn % Np);
  int kt = (int)(cn / Np);
  int k = kt * 32 + g * 16 + kk;
  float v = (k < K && n < N) ? W[(size_t)k * N + n] : 0.0f;
  out[idx] = (_Float16)v;
}

// ---------------------------------------------------------------------------
// Input gathers
// ---------------------------------------------------------------------------
__global__ void build_x_flight_kernel(const float* __restrict__ state,
                                      _Float16* __restrict__ Xf) {
  int r = blockIdx.x * blockDim.x + threadIdx.x;
  if (r >= BATCH * NF) return;
  int b = r / NF, f = r % NF;
  const float* srow = state + (size_t)b * ((NA + ND) * NF + NF);
  _Float16* xr = Xf + (size_t)r * 96;
#pragma unroll
  for (int j = 0; j < 64; ++j) xr[j] = (_Float16)srow[f * (NA + ND) + NA + j];
  xr[64] = (_Float16)srow[(NA + ND) * NF + f];
#pragma unroll
  for (int j = 65; j < 96; ++j) xr[j] = (_Float16)0.0f;
}

__global__ void build_x_act_kernel(const float* __restrict__ actions,
                                   _Float16* __restrict__ Xa) {
  int r = blockIdx.x * blockDim.x + threadIdx.x;
  if (r >= BATCH) return;
  const float* arow = actions + (size_t)r * (NA + ND);
  _Float16* xr = Xa + (size_t)r * 64;
#pragma unroll
  for (int j = 0; j < 64; ++j) xr[j] = (_Float16)arow[NA + j];
}

// ---------------------------------------------------------------------------
// Fused 3-layer MLP: out20 = ((relu(X@W1))@W2 relu)@W3, K0 -> 512 -> 512 -> 20
// 128 rows / workgroup, 8 waves as 4(M: 32 rows)x2(N: 256 cols); each wave
// holds two A fragments per B fragment (2x weight-stream reuse vs 64-row tile).
// ---------------------------------------------------------------------------
__global__ __launch_bounds__(256) void mlp3_kernel(
    const _Float16* __restrict__ X, int k0,              // k0 in {64, 96}
    const _Float16* __restrict__ W1p,                    // (k0 x 512) packed
    const _Float16* __restrict__ W2p,                    // (512 x 512) packed
    const _Float16* __restrict__ W3p,                    // (512 x 32) packed
    float* __restrict__ out20)                           // (M x 20) f32
{
  extern __shared__ _Float16 g_lds[];
  const int tid  = threadIdx.x;
  const int lane = tid & 31, w = tid >> 5;
  const int lm = lane & 15, lh = lane >> 4;
  const int m0 = (w & 3) * 32;
  const int wn = w >> 2;
  const size_t bm = (size_t)blockIdx.x * 128;

  _Float16* sX  = g_lds;                 // 128 x k0 (unskewed: TDM tile)
  _Float16* sH1 = sX + 128 * k0;         // 128 x HSTR
  _Float16* sH2 = sH1 + 128 * HSTR;      // 128 x HSTR

  const v8f z8 = {0.f, 0.f, 0.f, 0.f, 0.f, 0.f, 0.f, 0.f};

  // ---- stage X tile into LDS ----
#if HAVE_TDM
  if (w == 0) {
    tdm_load_2d(X + bm * (size_t)k0,
                (unsigned)(unsigned long long)(size_t)sX,
                (unsigned)k0, 128u, (unsigned)k0);
    __builtin_amdgcn_s_wait_tensorcnt(0);
  }
#else
  {
    const int nvec = (128 * k0) / 8;
    for (int i = tid; i < nvec; i += 256) {
      int r = i / (k0 / 8);
      int c = (i % (k0 / 8)) * 8;
      *(v8h*)(sX + r * k0 + c) = *(const v8h*)(X + (bm + r) * (size_t)k0 + c);
    }
  }
#endif
  __syncthreads();

  // ---- stage 1: H1 = relu(X @ W1), N = 512 ----
  const int kt0 = k0 >> 5;
  for (int nc = 0; nc < 4; ++nc) {
    int nb = wn * 256 + nc * 64;
    v8f acc[2][4] = {{z8, z8, z8, z8}, {z8, z8, z8, z8}};
    for (int kt = 0; kt < kt0; ++kt) {
      v16h a0 = load_a(sX + (m0 + lm) * k0 + kt * 32, lh);
      v16h a1 = load_a(sX + (m0 + 16 + lm) * k0 + kt * 32, lh);
#pragma unroll
      for (int j = 0; j < 4; ++j) {
        v16h b = load_b(W1p, 512, kt, nb + j * 16 + lm, lh);
        acc[0][j] = wmma_f16(a0, b, acc[0][j]);
        acc[1][j] = wmma_f16(a1, b, acc[1][j]);
      }
    }
#pragma unroll
    for (int s = 0; s < 2; ++s)
#pragma unroll
      for (int j = 0; j < 4; ++j)
#pragma unroll
        for (int r = 0; r < 8; ++r) {
          int row = m0 + s * 16 + lh * 8 + r;
          int col = nb + j * 16 + lm;
          float v = acc[s][j][r];
          sH1[row * HSTR + col] = (_Float16)(v > 0.f ? v : 0.f);
        }
  }
  __syncthreads();

  // ---- stage 2: H2 = relu(H1 @ W2), K = N = 512 ----
  for (int nc = 0; nc < 4; ++nc) {
    int nb = wn * 256 + nc * 64;
    v8f acc[2][4] = {{z8, z8, z8, z8}, {z8, z8, z8, z8}};
    for (int kt = 0; kt < 16; ++kt) {
      v16h a0 = load_a(sH1 + (m0 + lm) * HSTR + kt * 32, lh);
      v16h a1 = load_a(sH1 + (m0 + 16 + lm) * HSTR + kt * 32, lh);
      if (kt < 15)  // prefetch next kt's weight stream into L1/L2
        __builtin_prefetch(b_ptr(W2p, 512, kt + 1, nb + lm, lh), 0, 1);
#pragma unroll
      for (int j = 0; j < 4; ++j) {
        v16h b = load_b(W2p, 512, kt, nb + j * 16 + lm, lh);
        acc[0][j] = wmma_f16(a0, b, acc[0][j]);
        acc[1][j] = wmma_f16(a1, b, acc[1][j]);
      }
    }
#pragma unroll
    for (int s = 0; s < 2; ++s)
#pragma unroll
      for (int j = 0; j < 4; ++j)
#pragma unroll
        for (int r = 0; r < 8; ++r) {
          int row = m0 + s * 16 + lh * 8 + r;
          int col = nb + j * 16 + lm;
          float v = acc[s][j][r];
          sH2[row * HSTR + col] = (_Float16)(v > 0.f ? v : 0.f);
        }
  }
  __syncthreads();

  // ---- stage 3: out = H2 @ W3 (N padded 32, valid 20, no relu, f32 out) ----
  {
    int n0 = wn * 16;
    v8f acc0 = z8, acc1 = z8;
    for (int kt = 0; kt < 16; ++kt) {
      v16h a0 = load_a(sH2 + (m0 + lm) * HSTR + kt * 32, lh);
      v16h a1 = load_a(sH2 + (m0 + 16 + lm) * HSTR + kt * 32, lh);
      v16h b = load_b(W3p, 32, kt, n0 + lm, lh);
      acc0 = wmma_f16(a0, b, acc0);
      acc1 = wmma_f16(a1, b, acc1);
    }
    int col = n0 + lm;
    if (col < NF) {
#pragma unroll
      for (int r = 0; r < 8; ++r) {
        size_t row0 = bm + m0 + lh * 8 + r;
        out20[row0 * NF + col] = acc0[r];
        out20[(row0 + 16) * NF + col] = acc1[r];
      }
    }
  }
}

// ---------------------------------------------------------------------------
// Generic 64x64 WMMA GEMM tile with bias+relu, f16 out (value head layers).
// ---------------------------------------------------------------------------
__global__ __launch_bounds__(256) void gemm64_kernel(
    const _Float16* __restrict__ X, int ldx, int ktiles,
    const _Float16* __restrict__ Wp, int N,
    const float* __restrict__ bias, int nvalid, int do_relu,
    _Float16* __restrict__ out, int ldo)
{
  const int tid  = threadIdx.x;
  const int lane = tid & 31, w = tid >> 5;
  const int lm = lane & 15, lh = lane >> 4;
  const size_t bm = (size_t)blockIdx.x * 64;
  const int bn = blockIdx.y * 64;
  const int m0 = (w & 3) * 16;
  const int n0 = bn + (w >> 2) * 32;

  const v8f z8 = {0.f, 0.f, 0.f, 0.f, 0.f, 0.f, 0.f, 0.f};
  v8f acc0 = z8, acc1 = z8;

  const _Float16* xrow = X + (bm + m0 + lm) * (size_t)ldx;
  for (int kt = 0; kt < ktiles; ++kt) {
    v16h a = load_a(xrow + kt * 32, lh);
    if (kt + 1 < ktiles)
      __builtin_prefetch(b_ptr(Wp, N, kt + 1, n0 + lm, lh), 0, 1);
    v16h b0 = load_b(Wp, N, kt, n0 + lm, lh);
    v16h b1 = load_b(Wp, N, kt, n0 + 16 + lm, lh);
    acc0 = wmma_f16(a, b0, acc0);
    acc1 = wmma_f16(a, b1, acc1);
  }
#pragma unroll
  for (int j = 0; j < 2; ++j) {
    v8f acc = j ? acc1 : acc0;
    int col = n0 + j * 16 + lm;
    float bb = (bias != nullptr && col < nvalid) ? bias[col] : 0.0f;
#pragma unroll
    for (int r = 0; r < 8; ++r) {
      size_t row = bm + m0 + lh * 8 + r;
      float v = acc[r] + bb;
      if (do_relu && v < 0.f) v = 0.f;
      out[row * ldo + col] = (_Float16)v;
    }
  }
}

// ---------------------------------------------------------------------------
// Per-batch-item attention/layernorm block. One wave (32 lanes) per item.
// ---------------------------------------------------------------------------
__global__ __launch_bounds__(32) void attn_kernel(
    const float* __restrict__ emb20, const float* __restrict__ ea20,
    const float* __restrict__ Wk, const float* __restrict__ Wq,
    const float* __restrict__ Wv,
    const float* __restrict__ A1, const float* __restrict__ b1,
    const float* __restrict__ A2, const float* __restrict__ b2,
    const float* __restrict__ WkT, const float* __restrict__ WqT,
    const float* __restrict__ WvT,
    const float* __restrict__ C1, const float* __restrict__ c1,
    const float* __restrict__ C2, const float* __restrict__ c2,
    _Float16* __restrict__ post32)
{
  __shared__ float sK[NF], sQ[NF], sV[NF], sVec[NF], sAttn[NF * NF];
  __shared__ float sK2[NF], sQ2[NF], sAttn2[NF * NF];
  __shared__ float sEa[NF], sV2[NF], sAsa[NF], sAdd2[NF], sT[NF], sP[NF];

  const int b = blockIdx.x;
  const int f = threadIdx.x;
  float e[NF];

  if (f < NF) {
    const float* er = emb20 + ((size_t)b * NF + f) * NF;
    float k = 0.f, q = 0.f, v = 0.f;
#pragma unroll
    for (int j = 0; j < NF; ++j) {
      e[j] = er[j];
      k += e[j] * Wk[j];
      q += e[j] * Wq[j];
      v += e[j] * Wv[j];
    }
    sK[f] = k; sQ[f] = q; sV[f] = v;
  }
  __syncthreads();

  // attn = softmax over f (column-wise); lane g owns column g
  if (f < NF) {
    float qg = sQ[f];
    float m = -1e30f;
#pragma unroll
    for (int j = 0; j < NF; ++j) { float s = sK[j] * qg; m = s > m ? s : m; }
    float col[NF], den = 0.f;
#pragma unroll
    for (int j = 0; j < NF; ++j) { col[j] = __expf(sK[j] * qg - m); den += col[j]; }
    float inv = 1.f / den;
#pragma unroll
    for (int j = 0; j < NF; ++j) sAttn[j * NF + f] = col[j] * inv;
  }
  __syncthreads();

  if (f < NF) {
    float s = 0.f;
#pragma unroll
    for (int g = 0; g < NF; ++g) s += sAttn[f * NF + g] * sV[g];
    sVec[f] = s;
  }
  __syncthreads();

  // add = l2norm(sa + emb); ff; second = l2norm(ff + add); k2,q2
  if (f < NF) {
    float u[NF], n2 = 0.f;
#pragma unroll
    for (int j = 0; j < NF; ++j) { u[j] = sVec[j] + e[j]; n2 += u[j] * u[j]; }
    float inv = 1.f / fmaxf(sqrtf(n2), 1e-12f);
#pragma unroll
    for (int j = 0; j < NF; ++j) u[j] *= inv;
    float t[NF];
#pragma unroll
    for (int jj = 0; jj < NF; ++jj) {
      float s = b1[jj];
      for (int j = 0; j < NF; ++j) s += u[j] * A1[j * NF + jj];
      t[jj] = fmaxf(s, 0.f);
    }
    float sec[NF]; n2 = 0.f;
#pragma unroll
    for (int jj = 0; jj < NF; ++jj) {
      float s = b2[jj];
      for (int j = 0; j < NF; ++j) s += t[j] * A2[j * NF + jj];
      s += u[jj];
      sec[jj] = s; n2 += s * s;
    }
    inv = 1.f / fmaxf(sqrtf(n2), 1e-12f);
    float k2 = 0.f, q2 = 0.f;
#pragma unroll
    for (int j = 0; j < NF; ++j) { sec[j] *= inv; k2 += sec[j] * WkT[j]; q2 += sec[j] * WqT[j]; }
    sK2[f] = k2; sQ2[f] = q2;
  }
  __syncthreads();

  if (f < NF) {
    float qg = sQ2[f];
    float m = -1e30f;
#pragma unroll
    for (int j = 0; j < NF; ++j) { float s = sK2[j] * qg; m = s > m ? s : m; }
    float col[NF], den = 0.f;
#pragma unroll
    for (int j = 0; j < NF; ++j) { col[j] = __expf(sK2[j] * qg - m); den += col[j]; }
    float inv = 1.f / den;
#pragma unroll
    for (int j = 0; j < NF; ++j) sAttn2[j * NF + f] = col[j] * inv;
    sEa[f] = ea20[(size_t)b * NF + f];
  }
  __syncthreads();

  if (f < NF) {
    float s = 0.f;
#pragma unroll
    for (int j = 0; j < NF; ++j) s += sEa[j] * WvT[j * NF + f];
    sV2[f] = s;
  }
  __syncthreads();

  if (f < NF) {
    float s = 0.f;
#pragma unroll
    for (int g = 0; g < NF; ++g) s += sAttn2[f * NF + g] * sV2[g];
    sAsa[f] = s;
  }
  __syncthreads();

  {
    float n2 = 0.f;
#pragma unroll
    for (int j = 0; j < NF; ++j) { float u = sAsa[j] + sEa[j]; n2 += u * u; }
    float inv = 1.f / fmaxf(sqrtf(n2), 1e-12f);
    if (f < NF) sAdd2[f] = (sAsa[f] + sEa[f]) * inv;
  }
  __syncthreads();

  if (f < NF) {
    float s = c1[f];
#pragma unroll
    for (int j = 0; j < NF; ++j) s += sAdd2[j] * C1[j * NF + f];
    sT[f] = fmaxf(s, 0.f);
  }
  __syncthreads();

  if (f < NF) {
    float s = c2[f] + sAdd2[f];
#pragma unroll
    for (int j = 0; j < NF; ++j) s += sT[j] * C2[j * NF + f];
    sP[f] = s;
  }
  __syncthreads();

  {
    float n2 = 0.f;
#pragma unroll
    for (int j = 0; j < NF; ++j) n2 += sP[j] * sP[j];
    float inv = 1.f / fmaxf(sqrtf(n2), 1e-12f);
    _Float16 outv = (_Float16)0.0f;
    if (f < NF) outv = (_Float16)(sP[f] * inv);
    post32[(size_t)b * 32 + f] = outv;   // K-padded to 32 for the V1 GEMM
  }
}

// ---------------------------------------------------------------------------
// Tail: h4 = relu(h3 @ V4 + vb4); out = h4 @ V5 + vb5
// ---------------------------------------------------------------------------
__global__ void head_final_kernel(const _Float16* __restrict__ h3,
                                  const float* __restrict__ V4,
                                  const float* __restrict__ vb4,
                                  const float* __restrict__ V5,
                                  const float* __restrict__ vb5,
                                  float* __restrict__ out) {
  int r = blockIdx.x * blockDim.x + threadIdx.x;
  if (r >= BATCH) return;
  const _Float16* hr = h3 + (size_t)r * 64;
  float h[16];
#pragma unroll
  for (int j = 0; j < 16; ++j) h[j] = (float)hr[j];
  float s5 = vb5[0];
#pragma unroll
  for (int jj = 0; jj < 16; ++jj) {
    float s = vb4[jj];
#pragma unroll
    for (int j = 0; j < 16; ++j) s += h[j] * V4[j * 16 + jj];
    s = fmaxf(s, 0.f);
    s5 += s * V5[jj];
  }
  out[r] = s5;
}

// ---------------------------------------------------------------------------
// Host launcher
// ---------------------------------------------------------------------------
extern "C" void kernel_launch(void* const* d_in, const int* in_sizes, int n_in,
                              void* d_out, int out_size, void* d_ws, size_t ws_size,
                              hipStream_t stream) {
  const float* state   = (const float*)d_in[0];
  const float* actions = (const float*)d_in[1];
  const float* SE1 = (const float*)d_in[5];
  const float* SE2 = (const float*)d_in[6];
  const float* SE3 = (const float*)d_in[7];
  const float* Wk  = (const float*)d_in[8];
  const float* Wq  = (const float*)d_in[9];
  const float* Wv  = (const float*)d_in[10];
  const float* A1  = (const float*)d_in[11];
  const float* b1  = (const float*)d_in[12];
  const float* A2  = (const float*)d_in[13];
  const float* b2  = (const float*)d_in[14];
  const float* WkT = (const float*)d_in[15];
  const float* WqT = (const float*)d_in[16];
  const float* WvT = (const float*)d_in[17];
  const float* AE1 = (const float*)d_in[18];
  const float* AE2 = (const float*)d_in[19];
  const float* AE3 = (const float*)d_in[20];
  const float* C1  = (const float*)d_in[21];
  const float* c1  = (const float*)d_in[22];
  const float* C2  = (const float*)d_in[23];
  const float* c2  = (const float*)d_in[24];
  const float* V1  = (const float*)d_in[25];
  const float* vb1 = (const float*)d_in[26];
  const float* V2  = (const float*)d_in[27];
  const float* vb2 = (const float*)d_in[28];
  const float* V3  = (const float*)d_in[29];
  const float* vb3 = (const float*)d_in[30];
  const float* V4  = (const float*)d_in[31];
  const float* vb4 = (const float*)d_in[32];
  const float* V5  = (const float*)d_in[33];
  const float* vb5 = (const float*)d_in[34];

  // ---- workspace layout (256B aligned) ----
  char* ws = (char*)d_ws;
  size_t o = 0;
  auto alloc = [&](size_t bytes) -> char* {
    o = (o + 255) & ~(size_t)255;
    char* p = ws + o;
    o += bytes;
    return p;
  };
  _Float16* Xf     = (_Float16*)alloc((size_t)BATCH * NF * 96 * 2);
  _Float16* Xa     = (_Float16*)alloc((size_t)BATCH * 64 * 2);
  float*    emb20  = (float*)   alloc((size_t)BATCH * NF * NF * 4);
  float*    ea20   = (float*)   alloc((size_t)BATCH * NF * 4);
  _Float16* post32 = (_Float16*)alloc((size_t)BATCH * 32 * 2);
  _Float16* h1     = (_Float16*)alloc((size_t)BATCH * 1024 * 2);
  _Float16* h2     = (_Float16*)alloc((size_t)BATCH * 512 * 2);
  _Float16* h3     = (_Float16*)alloc((size_t)BATCH * 64 * 2);
  _Float16* SE1p   = (_Float16*)alloc((size_t)96 * 512 * 2);
  _Float16* SE2p   = (_Float16*)alloc((size_t)512 * 512 * 2);
  _Float16* SE3p   = (_Float16*)alloc((size_t)512 * 32 * 2);
  _Float16* AE1p   = (_Float16*)alloc((size_t)64 * 512 * 2);
  _Float16* AE2p   = (_Float16*)alloc((size_t)512 * 512 * 2);
  _Float16* AE3p   = (_Float16*)alloc((size_t)512 * 32 * 2);
  _Float16* V1p    = (_Float16*)alloc((size_t)32 * 1024 * 2);
  _Float16* V2p    = (_Float16*)alloc((size_t)1024 * 512 * 2);
  _Float16* V3p    = (_Float16*)alloc((size_t)512 * 64 * 2);
  (void)ws_size; (void)in_sizes; (void)n_in; (void)out_size;

  auto pack = [&](const float* W, int K, int N, int Kp, int Np, _Float16* out) {
    size_t total = (size_t)(Kp / 32) * Np * 32;
    int blocks = (int)((total + 255) / 256);
    pack_w_kernel<<<blocks, 256, 0, stream>>>(W, K, N, Kp, Np, out);
  };
  pack(SE1, 65, 512, 96, 512, SE1p);
  pack(SE2, 512, 512, 512, 512, SE2p);
  pack(SE3, 512, 20, 512, 32, SE3p);
  pack(AE1, 64, 512, 64, 512, AE1p);
  pack(AE2, 512, 512, 512, 512, AE2p);
  pack(AE3, 512, 20, 512, 32, AE3p);
  pack(V1, 20, 1024, 32, 1024, V1p);
  pack(V2, 1024, 512, 1024, 512, V2p);
  pack(V3, 512, 16, 512, 64, V3p);

  build_x_flight_kernel<<<(BATCH * NF + 255) / 256, 256, 0, stream>>>(state, Xf);
  build_x_act_kernel<<<(BATCH + 255) / 256, 256, 0, stream>>>(actions, Xa);

  // fused encoders: 128-row tiles, dyn LDS = 128*k0 + 2*128*520 halves
  size_t smemF = (size_t)(128 * 96 + 2 * 128 * HSTR) * 2;  // 290816 B
  size_t smemA = (size_t)(128 * 64 + 2 * 128 * HSTR) * 2;  // 282624 B
  (void)hipFuncSetAttribute((const void*)mlp3_kernel,
                            hipFuncAttributeMaxDynamicSharedMemorySize,
                            (int)smemF);
  mlp3_kernel<<<BATCH * NF / 128, 256, smemF, stream>>>(Xf, 96, SE1p, SE2p, SE3p, emb20);
  mlp3_kernel<<<BATCH / 128, 256, smemA, stream>>>(Xa, 64, AE1p, AE2p, AE3p, ea20);

  attn_kernel<<<BATCH, 32, 0, stream>>>(emb20, ea20, Wk, Wq, Wv, A1, b1, A2, b2,
                                        WkT, WqT, WvT, C1, c1, C2, c2, post32);

  gemm64_kernel<<<dim3(BATCH / 64, 1024 / 64), 256, 0, stream>>>(
      post32, 32, 1, V1p, 1024, vb1, 1024, 1, h1, 1024);
  gemm64_kernel<<<dim3(BATCH / 64, 512 / 64), 256, 0, stream>>>(
      h1, 1024, 32, V2p, 512, vb2, 512, 1, h2, 512);
  gemm64_kernel<<<dim3(BATCH / 64, 1), 256, 0, stream>>>(
      h2, 512, 16, V3p, 64, vb3, 16, 1, h3, 64);

  head_final_kernel<<<(BATCH + 255) / 256, 256, 0, stream>>>(h3, V4, vb4, V5, vb5,
                                                             (float*)d_out);
}
